// KernelAttention_65481071396711
// MI455X (gfx1250) — matrix operand order repaired and checked by reading, not compiled, not run on hardware
//
#include <hip/hip_runtime.h>

#define B_ 4
#define S_ 2048
#define E_ 512
#define NTOT (B_ * S_ * E_)

typedef __attribute__((ext_vector_type(16))) __bf16 v16bf;
typedef __attribute__((ext_vector_type(8)))  __bf16 v8bf;
typedef __attribute__((ext_vector_type(4)))  __bf16 v4bf;
typedef __attribute__((ext_vector_type(8)))  float  v8f;

static __device__ __forceinline__ __bf16 f2bf(float f) {
    unsigned u = __builtin_bit_cast(unsigned, f);
    unsigned r = (u + 0x7FFFu + ((u >> 16) & 1u)) >> 16;   // round-to-nearest-even
    unsigned short h = (unsigned short)r;
    return __builtin_bit_cast(__bf16, h);
}

static __device__ __forceinline__ v16bf cat16(v8bf a, v8bf b) {
    return __builtin_shufflevector(a, b, 0,1,2,3,4,5,6,7,8,9,10,11,12,13,14,15);
}

// ---------------------------------------------------------------------------
// Prep 1: LDS-tiled fp32 -> bf16 convert producing BOTH the straight copy
// xbf[b][s][e] and the transposed copy xT[b][e][s], with coalesced reads
// (float4) and coalesced 8B writes on both outputs.
// Tile = 32(s) x 32(e), 256 threads, 4 elems/thread.
// ---------------------------------------------------------------------------
__global__ __launch_bounds__(256) void prep_convert(const float* __restrict__ x,
                                                    __bf16* __restrict__ xbf,
                                                    __bf16* __restrict__ xT) {
    __shared__ __bf16 tile[32][33];      // +1 pad against bank conflicts

    const int tid = threadIdx.x;
    const int et  = blockIdx.x & 15;                 // E_/32 = 16
    const int st  = (blockIdx.x >> 4) & 63;          // S_/32 = 64
    const int b   = blockIdx.x >> 10;
    const int s0  = st * 32, e0 = et * 32;

    const int r  = tid >> 3;             // 0..31 (s within tile)
    const int c4 = (tid & 7) * 4;        // 0..28 (e within tile, x4)

    const float4 v = *(const float4*)(x + (size_t)((b * S_ + s0 + r) * E_) + e0 + c4);
    v4bf h;
    h[0] = f2bf(v.x); h[1] = f2bf(v.y); h[2] = f2bf(v.z); h[3] = f2bf(v.w);

    *(v4bf*)(xbf + (size_t)((b * S_ + s0 + r) * E_) + e0 + c4) = h;   // coalesced 8B
    tile[r][c4 + 0] = h[0]; tile[r][c4 + 1] = h[1];
    tile[r][c4 + 2] = h[2]; tile[r][c4 + 3] = h[3];
    __syncthreads();

    const int er  = tid >> 3;            // e within tile
    const int sc4 = (tid & 7) * 4;       // s within tile, x4
    v4bf t;
    t[0] = tile[sc4 + 0][er]; t[1] = tile[sc4 + 1][er];
    t[2] = tile[sc4 + 2][er]; t[3] = tile[sc4 + 3][er];
    *(v4bf*)(xT + (size_t)((b * E_ + e0 + er) * S_) + s0 + sc4) = t;  // coalesced 8B
}

// ---------------------------------------------------------------------------
// Prep 2: sq[b,s] = ||x_row||^2 (fp32).  One wave32 per row, shfl reduction.
// ---------------------------------------------------------------------------
__global__ __launch_bounds__(256) void prep_sq(const float* __restrict__ x,
                                               float* __restrict__ sq) {
    int lane = threadIdx.x & 31;
    int wv   = threadIdx.x >> 5;
    int row  = blockIdx.x * 8 + wv;          // flat b*S_ + s, 8192 rows total
    const float* p = x + (size_t)row * E_;
    float s = 0.f;
    #pragma unroll
    for (int k = 0; k < E_ / 32; ++k) { float v = p[lane + 32 * k]; s += v * v; }
    #pragma unroll
    for (int m = 16; m >= 1; m >>= 1) s += __shfl_xor(s, m, 32);
    if (lane == 0) sq[row] = s;
}

// ---------------------------------------------------------------------------
// Fused RBF attention.  Grid: B_ * S_/16 blocks of 256 threads (8 waves).
// Block owns 16 query rows; wave w owns output cols [w*64, w*64+64).
// softmax(K) with K = exp(-dist2) in (0,1] needs no running max: exp(K) <= e.
// ---------------------------------------------------------------------------
#define WPITCH 56   // bf16 elems per W row: 112B, 16B aligned, conflict-free

__global__ __launch_bounds__(256) void rbf_attn(const __bf16* __restrict__ xbf,
                                                const __bf16* __restrict__ xT,
                                                const float* __restrict__ sq,
                                                float* __restrict__ out) {
    __shared__ alignas(16) __bf16 Wl[8][16][WPITCH];  // 8 key-groups of 16x32 weights
    __shared__ float rs[8][16];                        // per-wave row-sum accumulators
    __shared__ float rstot[16];

    const int tid  = threadIdx.x;
    const int w    = tid >> 5;
    const int lane = tid & 31;
    const int hi   = lane >> 4;          // half-wave select
    const int ln   = lane & 15;
    const int b    = blockIdx.x >> 7;    // S_/16 = 128 query blocks per batch
    const int q0   = (blockIdx.x & 127) * 16;
    const int base = b * S_;
    const int cb   = w * 64;             // this wave's output column base

    if (tid < 128) rs[tid >> 4][tid & 15] = 0.f;
    __syncthreads();

    float sqq[8];
    #pragma unroll
    for (int v = 0; v < 8; ++v) sqq[v] = sq[base + q0 + v + 8 * hi];

    v8f o0 = {}, o1 = {}, o2 = {}, o3 = {};

    // A-fragment source rows for the score GEMM (per-lane query row)
    const __bf16* xq = xbf + (size_t)(base + q0 + ln) * E_;

    for (int kt = 0; kt < S_ / 256; ++kt) {
        const int kb = kt * 256 + w * 32;     // this wave's 32-key group
        // ---------------- scores: 16x32 tile, K = 512 ----------------
        v8f c0 = {}, c1 = {};
        const __bf16* bk0 = xbf + (size_t)(base + kb + ln) * E_ + hi * 16;
        const __bf16* bk1 = bk0 + (size_t)16 * E_;
        #pragma unroll
        for (int ks = 0; ks < 16; ++ks) {
            const int ko = ks * 32;
            v16bf a  = cat16(*(const v8bf*)(xq + ko + hi * 16),
                             *(const v8bf*)(xq + ko + hi * 16 + 8));
            v16bf bb0 = cat16(*(const v8bf*)(bk0 + ko), *(const v8bf*)(bk0 + ko + 8));
            v16bf bb1 = cat16(*(const v8bf*)(bk1 + ko), *(const v8bf*)(bk1 + ko + 8));
            c0 = __builtin_amdgcn_wmma_f32_16x16x32_bf16(false, a, false, bb0, (short)0, c0, false, false);
            c1 = __builtin_amdgcn_wmma_f32_16x16x32_bf16(false, a, false, bb1, (short)0, c1, false, false);
        }
        const float sqk0 = sq[base + kb + ln];
        const float sqk1 = sq[base + kb + 16 + ln];
        float rsum[8];
        #pragma unroll
        for (int v = 0; v < 8; ++v) {
            float d0 = fmaxf(sqq[v] + sqk0 - 2.f * c0[v], 0.f);
            float d1 = fmaxf(sqq[v] + sqk1 - 2.f * c1[v], 0.f);
            float w0 = __expf(__expf(-d0));
            float w1 = __expf(__expf(-d1));
            const int r = v + 8 * hi;
            Wl[w][r][ln]      = f2bf(w0);
            Wl[w][r][16 + ln] = f2bf(w1);
            rsum[v] = w0 + w1;
        }
        #pragma unroll
        for (int v = 0; v < 8; ++v) {        // deterministic row-sum reduction
            float s = rsum[v];
            s += __shfl_xor(s, 1, 32); s += __shfl_xor(s, 2, 32);
            s += __shfl_xor(s, 4, 32); s += __shfl_xor(s, 8, 32);
            if (ln == 0) rs[w][v + 8 * hi] += s;
        }
        __syncthreads();
        // ---------------- O += W(16x256) * X(256x64) ----------------
        const __bf16* xtb = xT + (size_t)(b * E_ + cb + ln) * S_ + kt * 256 + hi * 16;
        #pragma unroll
        for (int g = 0; g < 8; ++g) {
            const __bf16* wp = &Wl[g][ln][hi * 16];
            v16bf a = cat16(*(const v8bf*)(wp), *(const v8bf*)(wp + 8));
            const int ko = g * 32;
            v16bf bb0 = cat16(*(const v8bf*)(xtb + ko),                   *(const v8bf*)(xtb + ko + 8));
            v16bf bb1 = cat16(*(const v8bf*)(xtb + (size_t)16 * S_ + ko), *(const v8bf*)(xtb + (size_t)16 * S_ + ko + 8));
            v16bf bb2 = cat16(*(const v8bf*)(xtb + (size_t)32 * S_ + ko), *(const v8bf*)(xtb + (size_t)32 * S_ + ko + 8));
            v16bf bb3 = cat16(*(const v8bf*)(xtb + (size_t)48 * S_ + ko), *(const v8bf*)(xtb + (size_t)48 * S_ + ko + 8));
            o0 = __builtin_amdgcn_wmma_f32_16x16x32_bf16(false, a, false, bb0, (short)0, o0, false, false);
            o1 = __builtin_amdgcn_wmma_f32_16x16x32_bf16(false, a, false, bb1, (short)0, o1, false, false);
            o2 = __builtin_amdgcn_wmma_f32_16x16x32_bf16(false, a, false, bb2, (short)0, o2, false, false);
            o3 = __builtin_amdgcn_wmma_f32_16x16x32_bf16(false, a, false, bb3, (short)0, o3, false, false);
        }
        __syncthreads();                      // Wl reused next iteration
    }

    if (tid < 16) {
        float s = 0.f;
        #pragma unroll
        for (int g = 0; g < 8; ++g) s += rs[g][tid];
        rstot[tid] = s;
    }
    __syncthreads();

    #pragma unroll
    for (int v = 0; v < 8; ++v) {
        const int r = v + 8 * hi;
        const float inv = 1.f / rstot[r];
        float* op = out + (size_t)(base + q0 + r) * E_ + cb + ln;
        op[0]  = o0[v] * inv;
        op[16] = o1[v] * inv;
        op[32] = o2[v] * inv;
        op[48] = o3[v] * inv;
    }
}

// ---------------------------------------------------------------------------
extern "C" void kernel_launch(void* const* d_in, const int* in_sizes, int n_in,
                              void* d_out, int out_size, void* d_ws, size_t ws_size,
                              hipStream_t stream) {
    (void)in_sizes; (void)n_in; (void)out_size; (void)ws_size;
    const float* x = (const float*)d_in[0];
    float* out = (float*)d_out;

    // workspace layout: sq (B*S f32) | xbf16 (B*S*E) | xT bf16 (B*E*S)
    float*  sq  = (float*)d_ws;
    __bf16* xbf = (__bf16*)((char*)d_ws + (size_t)B_ * S_ * sizeof(float));
    __bf16* xT  = xbf + (size_t)NTOT;

    prep_convert<<<B_ * (S_ / 32) * (E_ / 32), 256, 0, stream>>>(x, xbf, xT);
    prep_sq<<<(B_ * S_) / 8, 256, 0, stream>>>(x, sq);
    rbf_attn<<<B_ * (S_ / 16), 256, 0, stream>>>(xbf, xT, sq, out);
}
